// GraphAttentionLayer_73821897883887
// MI455X (gfx1250) — matrix operand order repaired and checked by reading, not compiled, not run on hardware
//
#include <hip/hip_runtime.h>

// ---------------------------------------------------------------------------
// GAT layer, fused flash-attention style for MI455X (gfx1250, wave32, WMMA)
//   Phase 1: Wh = h @ W  (store f16 for WMMA B), Wh1 = Wh@a1, Wh2 = Wh@a2
//   Phase 2: out = elu( softmax(mask(leakyrelu(Wh1 + Wh2^T), adj)) @ Wh )
//            online softmax over j-tiles of 32, P@Wh via v_wmma_f32_16x16x32_f16
// ---------------------------------------------------------------------------

typedef __attribute__((ext_vector_type(16))) _Float16 v16h;
typedef __attribute__((ext_vector_type(8)))  _Float16 v8h;
typedef __attribute__((ext_vector_type(8)))  float    v8f;
typedef __attribute__((ext_vector_type(4)))  int      v4i;

#define N_NODES 8192
#define IN_F    128
#define OUT_F   64
#define ALPHA   0.2f
#define NEG_BIG (-9.0e15f)

#define WAVES   2
#define BM      (16 * WAVES)   // rows per block in phase 2
#define LDSJ    34             // padded LDS stride (halves): 17 dwords, odd -> bank-conflict-free

// ---------------------------------------------------------------------------
// Phase 1: Wh (f16), Wh1, Wh2
// block: 256 threads = 32 rows x 8 col-groups (8 cols each); grid: 256 blocks
// ---------------------------------------------------------------------------
__global__ __launch_bounds__(256) void gat_phase1(
    const float* __restrict__ h, const float* __restrict__ W,
    const float* __restrict__ a,
    _Float16* __restrict__ WhH, float* __restrict__ Wh1, float* __restrict__ Wh2)
{
    __shared__ float Wlds[IN_F * OUT_F];   // 32 KB
    __shared__ float aLds[2 * OUT_F];
    __shared__ float hLds[32 * IN_F];      // 16 KB

    const int tid = threadIdx.x;
    const int i0  = blockIdx.x * 32;

    for (int idx = tid; idx < IN_F * OUT_F; idx += 256) Wlds[idx] = W[idx];
    if (tid < 2 * OUT_F) aLds[tid] = a[tid];
    for (int idx = tid; idx < 32 * IN_F; idx += 256)
        hLds[idx] = h[(size_t)i0 * IN_F + idx];
    __syncthreads();

    const int r  = tid >> 3;      // local row 0..31
    const int cg = tid & 7;       // col group 0..7 (8 cols each)
    const int i  = i0 + r;

    float acc[8];
#pragma unroll
    for (int c = 0; c < 8; ++c) acc[c] = 0.f;

    for (int k = 0; k < IN_F; ++k) {
        const float hk = hLds[r * IN_F + k];
#pragma unroll
        for (int c = 0; c < 8; ++c)
            acc[c] += hk * Wlds[k * OUT_F + cg * 8 + c];
    }

    float p1 = 0.f, p2 = 0.f;
#pragma unroll
    for (int c = 0; c < 8; ++c) {
        const int col = cg * 8 + c;
        WhH[(size_t)i * OUT_F + col] = (_Float16)acc[c];
        p1 += acc[c] * aLds[col];
        p2 += acc[c] * aLds[OUT_F + col];
    }
    // reduce across the 8 lanes of this row (lanes are consecutive in wave32)
#pragma unroll
    for (int off = 1; off < 8; off <<= 1) {
        p1 += __shfl_xor(p1, off);
        p2 += __shfl_xor(p2, off);
    }
    if (cg == 0) { Wh1[i] = p1; Wh2[i] = p2; }
}

// ---------------------------------------------------------------------------
// Phase 2: fused masked-softmax-attention @ Wh, WMMA f16->f32
// block: 64 threads (2 waves), each wave owns 16 output rows; 256 j-tiles of 32
// ---------------------------------------------------------------------------
__global__ __launch_bounds__(32 * WAVES) void gat_phase2(
    const int* __restrict__ adj, const _Float16* __restrict__ WhH,
    const float* __restrict__ Wh1, const float* __restrict__ Wh2,
    float* __restrict__ out)
{
    __shared__ _Float16 whT[OUT_F * LDSJ];  // transposed Wh tile [col][j], padded
    __shared__ float    wh2t[32];

    const int tid  = threadIdx.x;
    const int lane = tid & 31;
    const int wave = tid >> 5;
    const int mrow = lane & 15;   // row (A frag) / column (B,C,D frags)
    const int kh   = lane >> 4;   // half-wave selector

    const int iRow = blockIdx.x * BM + wave * 16 + mrow;
    const float wh1i = Wh1[iRow];
    const int* adjRow = adj + (size_t)iRow * N_NODES;

    // A-fragment K slots owned by this lane (ISA 7.12.2, 16-bit A 16x32):
    //   lanes 0-15 : K = 0..7 and 16..23 ; lanes 16-31 : K = 8..15 and 24..31
    const int base1 = kh * 8;
    const int base2 = 16 + kh * 8;

    v8f acc[4] = {};              // 4 N-blocks of 16 cols; C/D layout: vgpr r = row r (+8 for kh=1)
    float m = -__builtin_inff();  // running row max
    float lpart = 0.f;            // this lane's partial row sum

    for (int jt = 0; jt < N_NODES / 32; ++jt) {
        const int j0 = jt * 32;
        __syncthreads();  // previous tile's WMMA reads of whT are done

        // --- stage Wh tile (32 x 64 f16) transposed into LDS ---
        for (int q = tid; q < (32 * OUT_F) / 8; q += 32 * WAVES) {
            const int j  = q >> 3;
            const int c8 = (q & 7) * 8;
            v8h chunk = *(const v8h*)&WhH[(size_t)(j0 + j) * OUT_F + c8];
#pragma unroll
            for (int c = 0; c < 8; ++c) whT[(c8 + c) * LDSJ + j] = chunk[c];
        }
        if (tid < 32) wh2t[tid] = Wh2[j0 + tid];
        __syncthreads();

        // --- adjacency: streamed once, nontemporal; prefetch next tile ---
        const int jpf = (j0 + 32 < N_NODES) ? (j0 + 32) : j0;
        __builtin_prefetch(adjRow + jpf + base1, 0, 0);
        v4i a0 = __builtin_nontemporal_load((const v4i*)(adjRow + j0 + base1));
        v4i a1 = __builtin_nontemporal_load((const v4i*)(adjRow + j0 + base1 + 4));
        v4i a2 = __builtin_nontemporal_load((const v4i*)(adjRow + j0 + base2));
        v4i a3 = __builtin_nontemporal_load((const v4i*)(adjRow + j0 + base2 + 4));
        const int av[16] = { a0.x,a0.y,a0.z,a0.w, a1.x,a1.y,a1.z,a1.w,
                             a2.x,a2.y,a2.z,a2.w, a3.x,a3.y,a3.z,a3.w };

        // --- scores: leaky-relu(Wh1[i]+Wh2[j]) masked to -9e15 (exact ref) ---
        float s[16];
#pragma unroll
        for (int t = 0; t < 16; ++t) {
            const int kk = (t < 8) ? (base1 + t) : (base2 + t - 8);
            float e = wh1i + wh2t[kk];
            e = (e > 0.f) ? e : ALPHA * e;
            s[t] = (av[t] > 0) ? e : NEG_BIG;
        }

        // --- online softmax update (row stats shared across lane pair L,L+16) ---
        float lm = s[0];
#pragma unroll
        for (int t = 1; t < 16; ++t) lm = fmaxf(lm, s[t]);
        const float cand  = fmaxf(lm, __shfl_xor(lm, 16));
        const float mnew  = fmaxf(m, cand);
        const float scale = __expf(m - mnew);   // 0 on first tile (m = -inf)

        union { v16h v; _Float16 hh[16]; } af;
        float psum = 0.f;
#pragma unroll
        for (int t = 0; t < 16; ++t) {
            const float p = __expf(s[t] - mnew);
            psum += p;
            af.hh[t] = (_Float16)p;
        }
        lpart = lpart * scale + psum;
        m = mnew;

        // rescale accumulators: row of acc vgpr r is (r + 8*kh); stats live on lane (row%16)
#pragma unroll
        for (int r = 0; r < 8; ++r) {
            const float sr = __shfl(scale, r + (kh << 3));
#pragma unroll
            for (int c = 0; c < 4; ++c) acc[c][r] *= sr;
        }

        // --- P (16x32 f16) @ Wh-tile (32x64 f16) via 4 WMMAs ---
#pragma unroll
        for (int c = 0; c < 4; ++c) {
            union { v16h v; unsigned u[8]; } bf;
            // B layout: lane half h: vgpr v<4 = B[8h+2v][n],B[8h+2v+1][n]; v>=4: K = 16+8h+2(v-4)
            const unsigned* bsrc =
                (const unsigned*)&whT[(c * 16 + mrow) * LDSJ + kh * 8];
#pragma unroll
            for (int v = 0; v < 8; ++v) bf.u[v] = bsrc[v < 4 ? v : v + 4];
            acc[c] = __builtin_amdgcn_wmma_f32_16x16x32_f16(
                false, af.v, false, bf.v, (short)0, acc[c], false, false);
        }
    }

    // --- finalize: normalize by row sum, ELU, store fp32 ---
    const float lrow = lpart + __shfl_xor(lpart, 16);
#pragma unroll
    for (int r = 0; r < 8; ++r) {
        const float lr  = __shfl(lrow, r + (kh << 3));
        const float inv = 1.f / lr;
        const int rowg  = blockIdx.x * BM + wave * 16 + r + (kh << 3);
#pragma unroll
        for (int c = 0; c < 4; ++c) {
            float x = acc[c][r] * inv;
            x = (x > 0.f) ? x : (__expf(x) - 1.f);
            out[(size_t)rowg * OUT_F + c * 16 + mrow] = x;
        }
    }
}

// ---------------------------------------------------------------------------
extern "C" void kernel_launch(void* const* d_in, const int* in_sizes, int n_in,
                              void* d_out, int out_size, void* d_ws, size_t ws_size,
                              hipStream_t stream) {
    const float* h   = (const float*)d_in[0];
    const int*   adj = (const int*)d_in[1];
    const float* W   = (const float*)d_in[2];
    const float* a   = (const float*)d_in[3];
    // d_in[4] = epoach (unused)

    // workspace layout: Wh f16 [N x OUT_F] | Wh1 f32 [N] | Wh2 f32 [N]
    _Float16* WhH = (_Float16*)d_ws;
    float*    Wh1 = (float*)((char*)d_ws + (size_t)N_NODES * OUT_F * sizeof(_Float16));
    float*    Wh2 = Wh1 + N_NODES;
    float*    out = (float*)d_out;

    gat_phase1<<<N_NODES / 32, 256, 0, stream>>>(h, W, a, WhH, Wh1, Wh2);
    gat_phase2<<<N_NODES / BM, 32 * WAVES, 0, stream>>>(adj, WhH, Wh1, Wh2, out);
}
